// BeliefTransformerBlock_8881992368729
// MI455X (gfx1250) — matrix operand order repaired and checked by reading, not compiled, not run on hardware
//
#include <hip/hip_runtime.h>

// ---------------------------------------------------------------------------
// CDNA5 (gfx1250) bf16-WMMA implementation of the BeliefTransformer block.
// All GEMMs use v_wmma_f32_16x16x32_bf16 (wave32). f32 accumulate everywhere;
// LN / softmax / GELU epilogues in f32.
// ---------------------------------------------------------------------------

typedef __attribute__((ext_vector_type(16))) __bf16 v16bf;
typedef __attribute__((ext_vector_type(8)))  float  v8f;

struct U8x32 { uint4 a, b; };   // 32 bytes == one 8-VGPR bf16 fragment

__device__ __forceinline__ unsigned short f2bf(float f) {
  unsigned int u = __float_as_uint(f);
  u += 0x7fffu + ((u >> 16) & 1u);          // round-to-nearest-even
  return (unsigned short)(u >> 16);
}

// A fragment, 16x32 bf16, from row-major [M,K]; p = &A[row(lane%16)][kk].
// ISA layout: K(v,half) = 16*(v>=4) + 8*half + 2*(v%4) + e  -> two contiguous
// 8-element (16B) loads per lane.
__device__ __forceinline__ v16bf ld_frag_a(const unsigned short* p, int half) {
  U8x32 u;
  u.a = *reinterpret_cast<const uint4*>(p + half * 8);
  u.b = *reinterpret_cast<const uint4*>(p + 16 + half * 8);
  return __builtin_bit_cast(v16bf, u);
}

// B fragment, 32x16 bf16, from row-major transposed operand [N,K];
// p = &Bt[col(lane%16)][kk]. ISA layout: K(v,half) = 16*(v>=4) + 8*half + 2*(v%4) + e
// (same as A) -> two contiguous 8-element (16B) loads per lane.
__device__ __forceinline__ v16bf ld_frag_b(const unsigned short* p, int half) {
  U8x32 u;
  u.a = *reinterpret_cast<const uint4*>(p + half * 8);
  u.b = *reinterpret_cast<const uint4*>(p + 16 + half * 8);
  return __builtin_bit_cast(v16bf, u);
}

__device__ __forceinline__ v8f wmma_bf16(v16bf a, v16bf b, v8f c) {
  return __builtin_amdgcn_wmma_f32_16x16x32_bf16(false, a, false, b,
                                                 (short)0, c, false, false);
}

// ---------------------------------------------------------------------------
// LayerNorm(f32 in) * gamma -> bf16 out. One wave per 512-element row.
// ---------------------------------------------------------------------------
__global__ __launch_bounds__(256)
void ln_cast_kernel(const float* __restrict__ x, const float* __restrict__ g,
                    unsigned short* __restrict__ out) {
  const int lane = threadIdx.x & 31;
  const int wave = threadIdx.x >> 5;
  const long row = (long)blockIdx.x * 8 + wave;

  const float4* x4 = reinterpret_cast<const float4*>(x + row * 512);
  float v[16];
#pragma unroll
  for (int i = 0; i < 4; ++i) {
    float4 t = x4[lane * 4 + i];
    v[4 * i + 0] = t.x; v[4 * i + 1] = t.y; v[4 * i + 2] = t.z; v[4 * i + 3] = t.w;
  }
  float s = 0.f, q = 0.f;
#pragma unroll
  for (int i = 0; i < 16; ++i) { s += v[i]; q += v[i] * v[i]; }
#pragma unroll
  for (int m = 16; m >= 1; m >>= 1) {
    s += __shfl_xor(s, m, 32);
    q += __shfl_xor(q, m, 32);
  }
  const float mu  = s * (1.f / 512.f);
  const float var = q * (1.f / 512.f) - mu * mu;
  const float rs  = rsqrtf(var + 1e-5f);

  const float4* g4 = reinterpret_cast<const float4*>(g);
  float gv[16];
#pragma unroll
  for (int i = 0; i < 4; ++i) {
    float4 t = g4[lane * 4 + i];
    gv[4 * i + 0] = t.x; gv[4 * i + 1] = t.y; gv[4 * i + 2] = t.z; gv[4 * i + 3] = t.w;
  }
  unsigned int pk[8];
#pragma unroll
  for (int i = 0; i < 8; ++i) {
    unsigned short lo = f2bf((v[2 * i]     - mu) * rs * gv[2 * i]);
    unsigned short hi = f2bf((v[2 * i + 1] - mu) * rs * gv[2 * i + 1]);
    pk[i] = (unsigned)lo | ((unsigned)hi << 16);
  }
  uint4* o4 = reinterpret_cast<uint4*>(out + row * 512);
  o4[lane * 2 + 0] = make_uint4(pk[0], pk[1], pk[2], pk[3]);
  o4[lane * 2 + 1] = make_uint4(pk[4], pk[5], pk[6], pk[7]);
}

// ---------------------------------------------------------------------------
// Weight transpose + cast: src f32 [R,C] -> dst bf16 [C,R].
// ---------------------------------------------------------------------------
__global__ __launch_bounds__(256)
void transpose_cast_kernel(const float* __restrict__ src,
                           unsigned short* __restrict__ dst, int R, int C) {
  long i = (long)blockIdx.x * 256 + threadIdx.x;
  if (i >= (long)R * C) return;
  int r = (int)(i / C), c = (int)(i % C);
  dst[(long)c * R + r] = f2bf(src[i]);
}

// ---------------------------------------------------------------------------
// Generic bf16 GEMM: C = A[M,K] @ Bt[N,K]^T.  Block tile 128x128 (8 waves,
// each 32(M)x64(N) = 2x4 WMMA tiles).  Epilogue selected by template.
// ---------------------------------------------------------------------------
enum { EP_F32 = 0, EP_GELU_BF16 = 2, EP_QKV = 3 };

template <int EP>
__global__ __launch_bounds__(256)
void gemm_bf16_kernel(const unsigned short* __restrict__ A,
                      const unsigned short* __restrict__ Bt,
                      void* __restrict__ Cv, void* __restrict__ C2,
                      void* __restrict__ C3,
                      int N, int K, long sA, long sB, long sC) {
  const int lane = threadIdx.x & 31;
  const int ln16 = lane & 15;
  const int half = lane >> 4;
  const int wave = threadIdx.x >> 5;
  const int wm = wave >> 1, wn = wave & 1;
  const long b = blockIdx.z;

  const unsigned short* Ab = A + b * sA;
  const unsigned short* Bb = Bt + b * sB;
  const int rowBase = blockIdx.y * 128 + wm * 32;
  const int colBase = blockIdx.x * 128 + wn * 64;

  const v8f vzero = {0.f, 0.f, 0.f, 0.f, 0.f, 0.f, 0.f, 0.f};
  v8f acc[2][4];
#pragma unroll
  for (int i = 0; i < 2; ++i)
#pragma unroll
    for (int j = 0; j < 4; ++j) acc[i][j] = vzero;

  const unsigned short* arow0 = Ab + (long)(rowBase + ln16) * K;
  const unsigned short* arow1 = Ab + (long)(rowBase + 16 + ln16) * K;
  const unsigned short* bbase = Bb + (long)(colBase + ln16) * K;
  const long ntStride = (long)16 * K;

  for (int kk = 0; kk < K; kk += 32) {
    v16bf a0 = ld_frag_a(arow0 + kk, half);
    v16bf a1 = ld_frag_a(arow1 + kk, half);
#pragma unroll
    for (int nt = 0; nt < 4; ++nt) {
      v16bf bf = ld_frag_b(bbase + nt * ntStride + kk, half);
      acc[0][nt] = wmma_bf16(a0, bf, acc[0][nt]);
      acc[1][nt] = wmma_bf16(a1, bf, acc[1][nt]);
    }
  }

#pragma unroll
  for (int mt = 0; mt < 2; ++mt)
#pragma unroll
    for (int nt = 0; nt < 4; ++nt)
#pragma unroll
      for (int v = 0; v < 8; ++v) {
        const int r = rowBase + mt * 16 + half * 8 + v;  // C layout: row=v+8*half
        const int c = colBase + nt * 16 + ln16;          //           col=lane%16
        const float val = acc[mt][nt][v];
        if (EP == EP_F32) {
          ((float*)Cv)[b * sC + (long)r * N + c] = val;
        } else if (EP == EP_GELU_BF16) {
          const float gv = 0.5f * val * (1.f + erff(val * 0.70710678118654752f));
          ((unsigned short*)Cv)[(long)r * N + c] = f2bf(gv);
        } else {  // EP_QKV: N=1536; split q,k row-major + v transposed [D,Ntok]
          if (c < 512) {
            ((unsigned short*)Cv)[(long)r * 512 + c] = f2bf(val);
          } else if (c < 1024) {
            ((unsigned short*)C2)[(long)r * 512 + (c - 512)] = f2bf(val);
          } else {
            const long bb = r >> 10;
            const int tok = r & 1023;
            const int d = c - 1024;
            ((unsigned short*)C3)[((bb << 9) + d) * 1024 + tok] = f2bf(val);
          }
        }
      }
}

// ---------------------------------------------------------------------------
// Fused scores + mask + softmax: block = (batch b, 32-query panel).
// 8 waves x 128-key chunks cover all 1024 keys; S held in registers.
// P (bf16 probabilities) written to ws; no f32 score tensor in HBM.
// ---------------------------------------------------------------------------
__global__ __launch_bounds__(256)
void attn_softmax_kernel(const unsigned short* __restrict__ q,
                         const unsigned short* __restrict__ k,
                         const int* __restrict__ sizes,
                         unsigned short* __restrict__ P) {
  __shared__ float s_red[8][32];
  __shared__ float s_row[32];

  const int lane = threadIdx.x & 31;
  const int ln16 = lane & 15;
  const int half = lane >> 4;
  const int wave = threadIdx.x >> 5;
  const int b = blockIdx.y;
  const int qb = blockIdx.x;
  const int vlen = sizes[b];

  const long qoff = ((long)b * 1024 + qb * 32) * 512;
  const unsigned short* qrow0 = q + qoff + (long)ln16 * 512;
  const unsigned short* qrow1 = q + qoff + (long)(16 + ln16) * 512;
  const int keyBase = wave * 128;
  const unsigned short* kbase =
      k + (long)b * 1024 * 512 + (long)(keyBase + ln16) * 512;

  const v8f vzero = {0.f, 0.f, 0.f, 0.f, 0.f, 0.f, 0.f, 0.f};
  v8f acc[2][8];
#pragma unroll
  for (int i = 0; i < 2; ++i)
#pragma unroll
    for (int j = 0; j < 8; ++j) acc[i][j] = vzero;

  for (int kk = 0; kk < 512; kk += 32) {
    v16bf a0 = ld_frag_a(qrow0 + kk, half);
    v16bf a1 = ld_frag_a(qrow1 + kk, half);
#pragma unroll
    for (int kt = 0; kt < 8; ++kt) {
      v16bf bf = ld_frag_b(kbase + kt * (16 * 512) + kk, half);
      acc[0][kt] = wmma_bf16(a0, bf, acc[0][kt]);
      acc[1][kt] = wmma_bf16(a1, bf, acc[1][kt]);
    }
  }

  const float scale = 0.044194173824159216f;  // 1/sqrt(512)
  bool valid[8];
#pragma unroll
  for (int kt = 0; kt < 8; ++kt) valid[kt] = (keyBase + kt * 16 + ln16) < vlen;

  // ---- per-row max (row = qt*16 + half*8 + v; lane%16 = key) ----
#pragma unroll
  for (int qt = 0; qt < 2; ++qt)
#pragma unroll
    for (int v = 0; v < 8; ++v) {
      float x = -3.0e38f;
#pragma unroll
      for (int kt = 0; kt < 8; ++kt)
        if (valid[kt]) x = fmaxf(x, acc[qt][kt][v] * scale);
#pragma unroll
      for (int m = 8; m >= 1; m >>= 1) x = fmaxf(x, __shfl_xor(x, m, 16));
      if (ln16 == 0) s_red[wave][qt * 16 + half * 8 + v] = x;
    }
  __syncthreads();
  if (threadIdx.x < 32) {
    float g = -3.0e38f;
#pragma unroll
    for (int w = 0; w < 8; ++w) g = fmaxf(g, s_red[w][threadIdx.x]);
    s_row[threadIdx.x] = g;
  }
  __syncthreads();

  // ---- exp + per-row sum ----
  float gm[2][8];
#pragma unroll
  for (int qt = 0; qt < 2; ++qt)
#pragma unroll
    for (int v = 0; v < 8; ++v) gm[qt][v] = s_row[qt * 16 + half * 8 + v];
  __syncthreads();  // s_red/s_row reused below

#pragma unroll
  for (int qt = 0; qt < 2; ++qt)
#pragma unroll
    for (int v = 0; v < 8; ++v) {
      float sum = 0.f;
#pragma unroll
      for (int kt = 0; kt < 8; ++kt) {
        float e = valid[kt] ? __expf(acc[qt][kt][v] * scale - gm[qt][v]) : 0.f;
        acc[qt][kt][v] = e;
        sum += e;
      }
#pragma unroll
      for (int m = 8; m >= 1; m >>= 1) sum += __shfl_xor(sum, m, 16);
      if (ln16 == 0) s_red[wave][qt * 16 + half * 8 + v] = sum;
    }
  __syncthreads();
  if (threadIdx.x < 32) {
    float g = 0.f;
#pragma unroll
    for (int w = 0; w < 8; ++w) g += s_red[w][threadIdx.x];
    s_row[threadIdx.x] = g;
  }
  __syncthreads();

  unsigned short* Pb = P + ((long)b * 1024 + qb * 32) * 1024;
#pragma unroll
  for (int qt = 0; qt < 2; ++qt)
#pragma unroll
    for (int v = 0; v < 8; ++v) {
      const int r = qt * 16 + half * 8 + v;
      const float rinv = 1.f / s_row[r];
#pragma unroll
      for (int kt = 0; kt < 8; ++kt)
        Pb[(long)r * 1024 + keyBase + kt * 16 + ln16] =
            f2bf(acc[qt][kt][v] * rinv);
    }
}

// ---------------------------------------------------------------------------
extern "C" void kernel_launch(void* const* d_in, const int* in_sizes, int n_in,
                              void* d_out, int out_size, void* d_ws,
                              size_t ws_size, hipStream_t stream) {
  const float* x      = (const float*)d_in[0];
  const int*   sizes  = (const int*)d_in[1];
  const float* g1     = (const float*)d_in[2];
  const float* w_qkv  = (const float*)d_in[3];
  const float* g2     = (const float*)d_in[4];
  const float* w_fc   = (const float*)d_in[5];
  const float* w_proj = (const float*)d_in[6];
  float* out = (float*)d_out;

  const long MB = 1ll << 20;
  char* w = (char*)d_ws;
  unsigned short* h      = (unsigned short*)(w);              // 32MB (LN1, reused LN2)
  unsigned short* qb     = (unsigned short*)(w + 32 * MB);    // 32MB q (reused fc_out)
  unsigned short* kb     = (unsigned short*)(w + 64 * MB);    // 32MB k
  unsigned short* vT     = (unsigned short*)(w + 96 * MB);    // 32MB v transposed [B,D,N]
  unsigned short* P      = (unsigned short*)(w + 128 * MB);   // 64MB probs bf16
  float*          y      = (float*)(w + 192 * MB);            // 64MB attention out f32
  unsigned short* wqkvT  = (unsigned short*)(w + 256 * MB);   // 1.5MB
  unsigned short* wfcT   = (unsigned short*)(w + 258 * MB);   // 0.5MB
  unsigned short* wprojT = (unsigned short*)(w + 259 * MB);   // 0.5MB

  // Weights -> bf16, transposed to [N,K]
  transpose_cast_kernel<<<(512 * 1536 + 255) / 256, 256, 0, stream>>>(w_qkv, wqkvT, 512, 1536);
  transpose_cast_kernel<<<(512 * 512 + 255) / 256, 256, 0, stream>>>(w_fc, wfcT, 512, 512);
  transpose_cast_kernel<<<(512 * 512 + 255) / 256, 256, 0, stream>>>(w_proj, wprojT, 512, 512);

  // LN1
  ln_cast_kernel<<<4096, 256, 0, stream>>>(x, g1, h);

  // qkv = h @ w_qkv  (M=32768, N=1536, K=512), split-store q/k/vT
  gemm_bf16_kernel<EP_QKV><<<dim3(12, 256, 1), 256, 0, stream>>>(
      h, wqkvT, qb, kb, vT, 1536, 512, 0, 0, 0);

  // masked softmax(q k^T / sqrt(D)) -> P
  attn_softmax_kernel<<<dim3(32, 32, 1), 256, 0, stream>>>(qb, kb, sizes, P);

  // y = P @ V  (batched: M=1024, N=512, K=1024)
  gemm_bf16_kernel<EP_F32><<<dim3(4, 8, 32), 256, 0, stream>>>(
      P, vT, y, nullptr, nullptr, 512, 1024,
      (long)1024 * 1024, (long)512 * 1024, (long)1024 * 512);

  // LN2
  ln_cast_kernel<<<4096, 256, 0, stream>>>(y, g2, h);

  // fc + exact GELU
  gemm_bf16_kernel<EP_GELU_BF16><<<dim3(4, 256, 1), 256, 0, stream>>>(
      h, wfcT, qb, nullptr, nullptr, 512, 512, 0, 0, 0);

  // proj -> f32 output
  gemm_bf16_kernel<EP_F32><<<dim3(4, 256, 1), 256, 0, stream>>>(
      qb, wprojT, out, nullptr, nullptr, 512, 512, 0, 0, 0);
}
